// CounterattackGNN_39994735460848
// MI455X (gfx1250) — matrix operand order repaired and checked by reading, not compiled, not run on hardware
//
#include <hip/hip_runtime.h>
#include <hip/hip_bf16.h>

typedef __attribute__((ext_vector_type(16))) __bf16 v16bf;
typedef __attribute__((ext_vector_type(8)))  __bf16 v8bf;
typedef __attribute__((ext_vector_type(8)))  float  v8f;

union AFrag { v16bf v; v8bf h[2]; };

__device__ __forceinline__ unsigned short f2bf(float f) {
  unsigned int u = __float_as_uint(f);
  unsigned int r = u + 0x7FFFu + ((u >> 16) & 1u);   // round-to-nearest-even
  return (unsigned short)(r >> 16);
}

// raw HW transcendentals: v_exp_f32 / v_log_f32 / v_rcp_f32, no fixup code
__device__ __forceinline__ float fast_exp(float x) {
  return __builtin_amdgcn_exp2f(x * 1.4426950408889634f);
}
__device__ __forceinline__ float sigmoidf_(float x) {
  return __builtin_amdgcn_rcpf(1.f + fast_exp(-x));
}
__device__ __forceinline__ float softplusf_(float x) {
  float t = __builtin_amdgcn_logf(1.f + fast_exp(-fabsf(x)));  // log2, arg in [1,2]
  return fmaxf(x, 0.f) + 0.69314718055994531f * t;
}

// ---------------------------------------------------------------------------
// Swizzle fp32 weights W[zdim][cout] into the bf16 B-fragment layout:
// out[((kb*ntiles + nt)*32 + lane)*16 + t], lane = (khalf<<4)|n,
// k = kb*32 + khalf*16 + t, c = nt*16 + n. 32B contiguous per lane.
// ---------------------------------------------------------------------------
__global__ void swizzle_weights(const float* __restrict__ W, unsigned short* __restrict__ out,
                                int zdim, int cout, int ntiles, int kchunks) {
  int id = blockIdx.x * blockDim.x + threadIdx.x;
  int total = kchunks * ntiles * 32 * 16;
  if (id >= total) return;
  int t    = id & 15;
  int lane = (id >> 4) & 31;
  int tile = id >> 9;
  int nt = tile % ntiles;
  int kb = tile / ntiles;
  int k = kb * 32 + (lane >> 4) * 16 + t;
  int c = nt * 16 + (lane & 15);
  float v = (k < zdim && c < cout) ? W[k * cout + c] : 0.f;
  out[id] = f2bf(v);
}

// ---------------------------------------------------------------------------
// Fused CGConv edge kernel.
//   CIN/COUT: channels (CIN % 4 == 0), KCH: K-chunks of 32 (pad of 2*CIN+6),
//   NT:  16-col output tiles (wave-parallel)
//   WET: wave-parallel 16-edge tiles           (NT*WET == 8 waves)
//   ESUB: sequential 16-edge tiles per wave, sharing B fragments
// Block covers ROWS = WET*ESUB*16 edges.
// ---------------------------------------------------------------------------
template<int CIN, int COUT, int KCH, int NT, int WET, int ESUB>
__global__ __launch_bounds__(256)
void cgconv_edge_wmma(const float* __restrict__ x,
                      const int* __restrict__ srcIdx,
                      const int* __restrict__ dstIdx,
                      const float* __restrict__ ea,
                      const unsigned short* __restrict__ wf_sw,
                      const unsigned short* __restrict__ ws_sw,
                      const float* __restrict__ bf,
                      const float* __restrict__ bs,
                      float* __restrict__ agg,
                      int E) {
  constexpr int KPAD = KCH * 32;
  constexpr int ROWS = WET * ESUB * 16;
  constexpr int SEGS_ROW = KPAD / 4;     // float4 segments per z row
  constexpr int SEG_X  = CIN / 4;        // segments holding x[dst]
  constexpr int SEG_2X = 2 * CIN / 4;    // first edge-attr segment
  __shared__ unsigned short zsm[ROWS * KPAD];
  __shared__ int s_dst[ROWS];
  __shared__ int s_src[ROWS];

  const int tid = threadIdx.x;
  const int eb = blockIdx.x * ROWS;
  const bool tail = (eb + ROWS > E);

  if (tid < ROWS) {
    int e = eb + tid;
    if (e < E) { s_dst[tid] = dstIdx[e]; s_src[tid] = srcIdx[e]; }
    else       { s_dst[tid] = -1;        s_src[tid] = 0; }
  }
  __syncthreads();

  // Gather + fp32->bf16 convert into LDS, float4 granularity.
  for (int sIdx = tid; sIdx < ROWS * SEGS_ROW; sIdx += 256) {
    int row = sIdx / SEGS_ROW;
    int seg = sIdx - row * SEGS_ROW;
    int d = s_dst[row];
    float4 v = make_float4(0.f, 0.f, 0.f, 0.f);
    if (!tail || d >= 0) {
      if (seg < SEG_X) {
        v = *reinterpret_cast<const float4*>(&x[(size_t)d * CIN + seg * 4]);
      } else if (seg < SEG_2X) {
        v = *reinterpret_cast<const float4*>(&x[(size_t)s_src[row] * CIN + (seg - SEG_X) * 4]);
      } else if (seg == SEG_2X) {
        const float* e = &ea[(size_t)(eb + row) * 6];
        v = make_float4(e[0], e[1], e[2], e[3]);
      } else if (seg == SEG_2X + 1) {
        const float* e = &ea[(size_t)(eb + row) * 6];
        v = make_float4(e[4], e[5], 0.f, 0.f);
      }
    }
    ushort4 o;
    o.x = f2bf(v.x); o.y = f2bf(v.y); o.z = f2bf(v.z); o.w = f2bf(v.w);
    *reinterpret_cast<ushort4*>(&zsm[row * KPAD + seg * 4]) = o;
  }
  __syncthreads();

  const int wave = tid >> 5;
  const int lane = tid & 31;
  const int etw  = wave / NT;        // wave-parallel edge-tile group
  const int nt   = wave - etw * NT;  // 16-col output tile owned by this wave
  const int mrow = lane & 15;
  const int sel  = lane >> 4;        // 0: K 0-15 ; 1: K 16-31

  const v16bf* bF = reinterpret_cast<const v16bf*>(wf_sw);
  const v16bf* bS = reinterpret_cast<const v16bf*>(ws_sw);

  v8f accF[ESUB];
  v8f accS[ESUB];
#pragma unroll
  for (int es = 0; es < ESUB; ++es) { accF[es] = v8f{}; accS[es] = v8f{}; }

#pragma unroll
  for (int kb = 0; kb < KCH; ++kb) {
    // B fragments: one 32B contiguous vector per lane, shared by all ESUB tiles
    v16bf fw = bF[(kb * NT + nt) * 32 + lane];
    v16bf sw = bS[(kb * NT + nt) * 32 + lane];
    if (kb + 1 < KCH) {
      __builtin_prefetch(&bF[((kb + 1) * NT + nt) * 32 + lane], 0, 0);
      __builtin_prefetch(&bS[((kb + 1) * NT + nt) * 32 + lane], 0, 0);
    }
#pragma unroll
    for (int es = 0; es < ESUB; ++es) {
      const int rowbase = (etw * ESUB + es) * 16;
      AFrag a;
      const unsigned short* zp =
          &zsm[(rowbase + mrow) * KPAD + kb * 32 + sel * 16];
      a.h[0] = *reinterpret_cast<const v8bf*>(zp);        // K 0-7  (or 16-23)
      a.h[1] = *reinterpret_cast<const v8bf*>(zp + 8);    // K 8-15 (or 24-31)
      accF[es] = __builtin_amdgcn_wmma_f32_16x16x32_bf16(false, a.v, false, fw,
                                                         (short)0, accF[es], false, false);
      accS[es] = __builtin_amdgcn_wmma_f32_16x16x32_bf16(false, a.v, false, sw,
                                                         (short)0, accS[es], false, false);
    }
  }

  // C-tile layout: lanes 0-15 -> M=r, lanes 16-31 -> M=r+8; N = lane&15
  const int col = nt * 16 + mrow;
  if (col < COUT) {
    const float bfv = bf[col];
    const float bsv = bs[col];
    if (!tail) {
#pragma unroll
      for (int es = 0; es < ESUB; ++es) {
        const int rowbase = (etw * ESUB + es) * 16;
#pragma unroll
        for (int r = 0; r < 8; ++r) {
          int d = s_dst[rowbase + r + (sel << 3)];
          float g = sigmoidf_(accF[es][r] + bfv);
          float s = softplusf_(accS[es][r] + bsv);
          atomicAdd(&agg[d * COUT + col], g * s);
        }
      }
    } else {
#pragma unroll
      for (int es = 0; es < ESUB; ++es) {
        const int rowbase = (etw * ESUB + es) * 16;
#pragma unroll
        for (int r = 0; r < 8; ++r) {
          int d = s_dst[rowbase + r + (sel << 3)];
          if (d >= 0) {
            float g = sigmoidf_(accF[es][r] + bfv);
            float s = softplusf_(accS[es][r] + bsv);
            atomicAdd(&agg[d * COUT + col], g * s);
          }
        }
      }
    }
  }
}

// out = relu(x + agg), elementwise
__global__ void add_relu_kernel(const float* __restrict__ x, const float* __restrict__ agg,
                                float* __restrict__ out, int n) {
  int i = blockIdx.x * blockDim.x + threadIdx.x;
  if (i < n) out[i] = fmaxf(x[i] + agg[i], 0.f);
}

// hA = h0 @ Wlin + blin   (12 -> 128)
__global__ void linear12_128(const float* __restrict__ h, const float* __restrict__ W,
                             const float* __restrict__ b, float* __restrict__ out, int nNodes) {
  int i = blockIdx.x * blockDim.x + threadIdx.x;
  if (i >= nNodes * 128) return;
  int node = i >> 7, c = i & 127;
  float acc = b[c];
#pragma unroll
  for (int k = 0; k < 12; ++k) acc += h[node * 12 + k] * W[k * 128 + c];
  out[i] = acc;
}

// segment sums + counts for global_mean_pool
__global__ void pool_kernel(const float* __restrict__ h, const int* __restrict__ batch,
                            float* __restrict__ sum, float* __restrict__ cnt, int nNodes) {
  int i = blockIdx.x * blockDim.x + threadIdx.x;
  if (i >= nNodes * 128) return;
  int node = i >> 7, c = i & 127;
  int b = batch[node];
  atomicAdd(&sum[b * 128 + c], h[i]);
  if (c == 0) atomicAdd(&cnt[b], 1.f);
}

// per-graph MLP head: relu(g@Wh1+bh1) -> relu(@Wh2+bh2) -> @Wh3+bh3
__global__ __launch_bounds__(128)
void head_kernel(const float* __restrict__ psum, const float* __restrict__ pcnt,
                 const float* __restrict__ Wh1, const float* __restrict__ bh1,
                 const float* __restrict__ Wh2, const float* __restrict__ bh2,
                 const float* __restrict__ Wh3, const float* __restrict__ bh3,
                 float* __restrict__ out) {
  __shared__ float gs[128];
  __shared__ float a1[128];
  __shared__ float a2[128];
  int b = blockIdx.x, t = threadIdx.x;
  float cnt = fmaxf(pcnt[b], 1.f);
  gs[t] = psum[b * 128 + t] / cnt;
  __syncthreads();
  float acc = bh1[t];
  for (int k = 0; k < 128; ++k) acc += gs[k] * Wh1[k * 128 + t];
  a1[t] = fmaxf(acc, 0.f);
  __syncthreads();
  acc = bh2[t];
  for (int k = 0; k < 128; ++k) acc += a1[k] * Wh2[k * 128 + t];
  a2[t] = fmaxf(acc, 0.f) * Wh3[t];
  __syncthreads();
  for (int s = 64; s > 0; s >>= 1) {
    if (t < s) a2[t] += a2[t + s];
    __syncthreads();
  }
  if (t == 0) out[b] = a2[0] + bh3[0];
}

extern "C" void kernel_launch(void* const* d_in, const int* in_sizes, int n_in,
                              void* d_out, int out_size, void* d_ws, size_t ws_size,
                              hipStream_t stream) {
  (void)in_sizes; (void)n_in; (void)out_size; (void)ws_size;

  const float* x    = (const float*)d_in[0];
  const int*   ei   = (const int*)d_in[1];
  const float* ea   = (const float*)d_in[2];
  const int*   batc = (const int*)d_in[3];
  const float* Wf1  = (const float*)d_in[4];  const float* bf1 = (const float*)d_in[5];
  const float* Ws1  = (const float*)d_in[6];  const float* bs1 = (const float*)d_in[7];
  const float* Wlin = (const float*)d_in[8];  const float* blin= (const float*)d_in[9];
  const float* Wf2  = (const float*)d_in[10]; const float* bf2 = (const float*)d_in[11];
  const float* Ws2  = (const float*)d_in[12]; const float* bs2 = (const float*)d_in[13];
  const float* Wf3  = (const float*)d_in[14]; const float* bf3 = (const float*)d_in[15];
  const float* Ws3  = (const float*)d_in[16]; const float* bs3 = (const float*)d_in[17];
  const float* Wh1  = (const float*)d_in[18]; const float* bh1 = (const float*)d_in[19];
  const float* Wh2  = (const float*)d_in[20]; const float* bh2 = (const float*)d_in[21];
  const float* Wh3  = (const float*)d_in[22]; const float* bh3 = (const float*)d_in[23];
  float* out = (float*)d_out;

  const int NN = 100000, NE = 600000, NG = 1000;
  const int* srcIdx = ei;          // edge_index[0]
  const int* dstIdx = ei + NE;     // edge_index[1]

  char* ws = (char*)d_ws;
  size_t off = 0;
  auto take = [&](size_t bytes) -> char* {
    char* p = ws + off;
    off = (off + bytes + 255) & ~(size_t)255;
    return p;
  };
  float* hA   = (float*)take((size_t)NN * 128 * 4);
  float* hB   = (float*)take((size_t)NN * 128 * 4);
  float* agg  = (float*)take((size_t)NN * 128 * 4);
  float* h0   = (float*)take((size_t)NN * 12 * 4);
  unsigned short* wf1s = (unsigned short*)take(1 * 1 * 512 * 2);
  unsigned short* ws1s = (unsigned short*)take(1 * 1 * 512 * 2);
  unsigned short* wf2s = (unsigned short*)take((size_t)9 * 8 * 512 * 2);
  unsigned short* ws2s = (unsigned short*)take((size_t)9 * 8 * 512 * 2);
  unsigned short* wf3s = (unsigned short*)take((size_t)9 * 8 * 512 * 2);
  unsigned short* ws3s = (unsigned short*)take((size_t)9 * 8 * 512 * 2);
  float* psum = (float*)take((size_t)NG * 128 * 4);
  float* pcnt = (float*)take((size_t)NG * 4);

  // --- weight swizzle (fp32 -> bf16 B-fragment layout) ---
  swizzle_weights<<<2, 256, 0, stream>>>(Wf1, wf1s, 30, 12, 1, 1);
  swizzle_weights<<<2, 256, 0, stream>>>(Ws1, ws1s, 30, 12, 1, 1);
  swizzle_weights<<<144, 256, 0, stream>>>(Wf2, wf2s, 262, 128, 8, 9);
  swizzle_weights<<<144, 256, 0, stream>>>(Ws2, ws2s, 262, 128, 8, 9);
  swizzle_weights<<<144, 256, 0, stream>>>(Wf3, wf3s, 262, 128, 8, 9);
  swizzle_weights<<<144, 256, 0, stream>>>(Ws3, ws3s, 262, 128, 8, 9);

  // --- layer 1: CGConv(C=12) + relu ---  (1 N-tile, 8 wave-parallel edge tiles)
  hipMemsetAsync(agg, 0, (size_t)NN * 12 * 4, stream);
  int blocks1 = (NE + 127) / 128;
  cgconv_edge_wmma<12, 12, 1, 1, 8, 1><<<blocks1, 256, 0, stream>>>(
      x, srcIdx, dstIdx, ea, wf1s, ws1s, bf1, bs1, agg, NE);
  add_relu_kernel<<<(NN * 12 + 255) / 256, 256, 0, stream>>>(x, agg, h0, NN * 12);

  // --- lift 12 -> 128 ---
  linear12_128<<<(NN * 128 + 255) / 256, 256, 0, stream>>>(h0, Wlin, blin, hA, NN);

  // --- layer 2: CGConv(C=128) + relu ---  (8 N-tiles, 4 sequential edge tiles)
  hipMemsetAsync(agg, 0, (size_t)NN * 128 * 4, stream);
  int blocks2 = (NE + 63) / 64;
  cgconv_edge_wmma<128, 128, 9, 8, 1, 4><<<blocks2, 256, 0, stream>>>(
      hA, srcIdx, dstIdx, ea, wf2s, ws2s, bf2, bs2, agg, NE);
  add_relu_kernel<<<(NN * 128 + 255) / 256, 256, 0, stream>>>(hA, agg, hB, NN * 128);

  // --- layer 3: CGConv(C=128) + relu ---
  hipMemsetAsync(agg, 0, (size_t)NN * 128 * 4, stream);
  cgconv_edge_wmma<128, 128, 9, 8, 1, 4><<<blocks2, 256, 0, stream>>>(
      hB, srcIdx, dstIdx, ea, wf3s, ws3s, bf3, bs3, agg, NE);
  add_relu_kernel<<<(NN * 128 + 255) / 256, 256, 0, stream>>>(hB, agg, hA, NN * 128);

  // --- global mean pool + head ---
  hipMemsetAsync(psum, 0, (size_t)NG * 128 * 4, stream);
  hipMemsetAsync(pcnt, 0, (size_t)NG * 4, stream);
  pool_kernel<<<(NN * 128 + 255) / 256, 256, 0, stream>>>(hA, batc, psum, pcnt, NN);
  head_kernel<<<NG, 128, 0, stream>>>(psum, pcnt, Wh1, bh1, Wh2, bh2, Wh3, bh3, out);
}
